// VQVAE_1640677507238
// MI455X (gfx1250) — matrix operand order repaired and checked by reading, not compiled, not run on hardware
//
#include <hip/hip_runtime.h>

// ---------------- types ----------------
typedef __attribute__((ext_vector_type(16))) __bf16 v16bf;
typedef __attribute__((ext_vector_type(8)))  __bf16 v8bf;
typedef __attribute__((ext_vector_type(8)))  float  v8f;

// ---------------- problem dims ----------------
#define NB    32
#define LEN   8192
#define TT    2048          // t = L/4
#define HID   64
#define CBD   512
#define NTOK  1024
#define NROW  65536         // B * t

// ---------------- d_out layout (floats) ----------------
#define DO_OUT    0          // 262144
#define DO_RECON  262144     // 1
#define DO_COMMIT 262145     // 1
#define DO_IDX    262146     // 65536
#define DO_QUANT  327682     // 33554432

// ---------------- workspace layout (bytes) ----------------
#define OFF_FLAT    0ull                    // 65536*512*4 = 134217728
#define OFF_FLATBF  134217728ull            // 65536*512*2 =  67108864
#define OFF_QBF     201326592ull            //                67108864
#define OFF_H1      268435456ull            // 32*64*4096*4 = 33554432
#define OFF_H2      301989888ull            // 32*64*2048*4 = 16777216
#define OFF_D1      318767104ull            //                33554432
#define OFF_CBBF    352321536ull            // 1024*512*2  =   1048576
#define OFF_CBN     353370112ull            // 1024*4
#define OFF_WA      353374208ull            // 2*64*512*2  =    131072
#define OFF_IDXI    353505280ull            // 65536*4
#define OFF_CPART   353767424ull            // 65536*4
#define OFF_RPART   354029568ull            // 1024*4

__device__ __forceinline__ unsigned short f2bf(float f) {
    unsigned int u = __float_as_uint(f);
    u += 0x7fffu + ((u >> 16) & 1u);        // round-to-nearest-even
    return (unsigned short)(u >> 16);
}

// ---------- prep: codebook -> bf16 + row norms ----------
__global__ __launch_bounds__(256) void vq_prep_cb(const float* __restrict__ cb,
                                                  unsigned short* __restrict__ cbbf,
                                                  float* __restrict__ cbn) {
    __shared__ float red[256];
    const int code = blockIdx.x;
    float s = 0.f;
    for (int c = threadIdx.x; c < CBD; c += 256) {
        float v = cb[(size_t)code * CBD + c];
        cbbf[(size_t)code * CBD + c] = f2bf(v);
        s += v * v;
    }
    red[threadIdx.x] = s; __syncthreads();
    for (int o = 128; o > 0; o >>= 1) { if (threadIdx.x < o) red[threadIdx.x] += red[threadIdx.x + o]; __syncthreads(); }
    if (threadIdx.x == 0) cbn[code] = red[0];
}

// ---------- prep: dec_w1 [512][64][2] -> wA[k][oc][ic] bf16 ----------
__global__ __launch_bounds__(256) void vq_prep_w(const float* __restrict__ dw1,
                                                 unsigned short* __restrict__ wA) {
    int id = blockIdx.x * 256 + threadIdx.x;        // 2*64*512 = 65536
    if (id >= 2 * HID * CBD) return;
    int ic = id & 511, oc = (id >> 9) & 63, k = id >> 15;
    wA[id] = f2bf(dw1[((size_t)ic * HID + oc) * 2 + k]);
}

// ---------- encoder stage 1: Conv1d(1,64,2,s2)+ReLU ----------
__global__ __launch_bounds__(256) void vq_enc1(const float* __restrict__ img,
                                               const float* __restrict__ w1,
                                               const float* __restrict__ b1,
                                               float* __restrict__ h1) {
    int id = blockIdx.x * 256 + threadIdx.x;        // [b][oc][i], i fastest
    int i = id & 4095, oc = (id >> 12) & 63, b = id >> 18;
    const float* ip = img + (size_t)b * LEN + 2 * i;
    float v = b1[oc] + w1[oc * 2] * ip[0] + w1[oc * 2 + 1] * ip[1];
    h1[id] = v > 0.f ? v : 0.f;
}

// ---------- encoder stage 2: Conv1d(64,64,2,s2)+ReLU ----------
__global__ __launch_bounds__(256) void vq_enc2(const float* __restrict__ h1,
                                               const float* __restrict__ w2,
                                               const float* __restrict__ b2,
                                               float* __restrict__ h2) {
    int i  = blockIdx.x * 256 + threadIdx.x;        // 0..2047
    int oc = blockIdx.y, b = blockIdx.z;
    const float* hb = h1 + (size_t)b * HID * 4096;
    const float* w  = w2 + oc * 128;
    float acc = b2[oc];
    #pragma unroll 8
    for (int ic = 0; ic < HID; ++ic) {
        const float* p = hb + ic * 4096 + 2 * i;
        acc += w[ic * 2] * p[0] + w[ic * 2 + 1] * p[1];
    }
    h2[((size_t)b * HID + oc) * TT + i] = acc > 0.f ? acc : 0.f;
}

// ---------- encoder stage 3: 1x1 conv 64->512, emit flat f32 + bf16 ----------
__global__ __launch_bounds__(256) void vq_enc3(const float* __restrict__ h2,
                                               const float* __restrict__ w3,
                                               const float* __restrict__ b3,
                                               float* __restrict__ flat,
                                               unsigned short* __restrict__ flatbf) {
    __shared__ float col[HID];
    const int r = blockIdx.x;                       // row = b*t + i
    const int b = r >> 11, i = r & 2047;
    if (threadIdx.x < HID) col[threadIdx.x] = h2[((size_t)b * HID + threadIdx.x) * TT + i];
    __syncthreads();
    const int c = blockIdx.y * 256 + threadIdx.x;   // 0..511
    const float* w = w3 + c * HID;
    float acc = b3[c];
    #pragma unroll
    for (int ic = 0; ic < HID; ++ic) acc += w[ic] * col[ic];
    flat  [(size_t)r * CBD + c] = acc;
    flatbf[(size_t)r * CBD + c] = f2bf(acc);
}

// ---------- VQ nearest-code search via WMMA bf16 ----------
#define SMSTRIDE 520   // 512 + 8 bf16 pad: row stride 1040 B -> 4-bank skew, 16-byte aligned
__global__ __launch_bounds__(256) void vq_argmin(const unsigned short* __restrict__ xbf,
                                                 const unsigned short* __restrict__ cbbf,
                                                 const float* __restrict__ cbn,
                                                 int* __restrict__ out_idx) {
    __shared__ unsigned short smA[128 * SMSTRIDE];  // 133 KB of 320 KB WGP LDS
    const int row0 = blockIdx.x * 128;
    // stage 128 rows x 512 bf16, 16B chunks
    for (int c = threadIdx.x; c < 128 * 64; c += 256) {
        int r = c >> 6, k8 = c & 63;
        uint4 v = *(const uint4*)(xbf + (size_t)(row0 + r) * CBD + k8 * 8);
        *(uint4*)(&smA[r * SMSTRIDE + k8 * 8]) = v;
    }
    __syncthreads();

    const int lane  = threadIdx.x & 31;
    const int wave  = threadIdx.x >> 5;
    const int mrow  = wave * 16;                    // wave's 16-row tile
    const int l16   = lane & 15;
    const int lhalf = lane >> 4;

    float minv[8]; int mini[8];
    #pragma unroll
    for (int v = 0; v < 8; ++v) { minv[v] = 3.4e38f; mini[v] = 0; }

    const unsigned short* smArow = &smA[(mrow + l16) * SMSTRIDE];
    for (int nt = 0; nt < NTOK / 16; ++nt) {
        const int mycode = nt * 16 + l16;           // N index held by this lane
        const unsigned short* brow = cbbf + (size_t)mycode * CBD + lhalf * 16;
        v8f acc = {};
        #pragma unroll
        for (int kt = 0; kt < 16; ++kt) {
            const int kb = kt * 32 + lhalf * 16;    // A: contiguous K layout (k = 16h + s), matches B
            v8bf alo = *(const v8bf*)(smArow + kb);
            v8bf ahi = *(const v8bf*)(smArow + kb + 8);
            v16bf a;
            #pragma unroll
            for (int q = 0; q < 8; ++q) { a[q] = alo[q]; a[8 + q] = ahi[q]; }
            v16bf bm = *(const v16bf*)(brow + kt * 32);
            acc = __builtin_amdgcn_wmma_f32_16x16x32_bf16(false, a, false, bm,
                                                          (short)0, acc, false, false);
        }
        const float cn = cbn[mycode];
        #pragma unroll
        for (int v = 0; v < 8; ++v) {
            float d = cn - 2.0f * acc[v];           // |e|^2 - 2 x.e  (|x|^2 dropped)
            if (d < minv[v]) { minv[v] = d; mini[v] = mycode; }
        }
    }
    // reduce across the 16 lanes holding each row (xor<=8 stays in half)
    #pragma unroll
    for (int off = 1; off < 16; off <<= 1) {
        #pragma unroll
        for (int v = 0; v < 8; ++v) {
            float ov = __shfl_xor(minv[v], off, 32);
            int   oi = __shfl_xor(mini[v], off, 32);
            if (ov < minv[v] || (ov == minv[v] && oi < mini[v])) { minv[v] = ov; mini[v] = oi; }
        }
    }
    if (l16 == 0) {
        #pragma unroll
        for (int v = 0; v < 8; ++v)
            out_idx[row0 + mrow + lhalf * 8 + v] = mini[v];
    }
}

// ---------- gather + quantized output + bf16 mirror + commit partials ----------
__global__ __launch_bounds__(256) void vq_gather(const int* __restrict__ idxbuf,
                                                 const float* __restrict__ cb,
                                                 const float* __restrict__ flat,
                                                 float* __restrict__ dout,
                                                 unsigned short* __restrict__ qbf,
                                                 float* __restrict__ cpart) {
    __shared__ float red[256];
    const int r = blockIdx.x, b = r >> 11, i = r & 2047;
    const int idx = idxbuf[r];
    float s = 0.f;
    float* qout = dout + DO_QUANT;
    for (int c = threadIdx.x; c < CBD; c += 256) {
        float q = cb[(size_t)idx * CBD + c];
        qout[((size_t)b * CBD + c) * TT + i] = q;   // [B][CBD][t]
        qbf[(size_t)r * CBD + c] = f2bf(q);
        float d = q - flat[(size_t)r * CBD + c];
        s += d * d;
    }
    red[threadIdx.x] = s; __syncthreads();
    for (int o = 128; o > 0; o >>= 1) { if (threadIdx.x < o) red[threadIdx.x] += red[threadIdx.x + o]; __syncthreads(); }
    if (threadIdx.x == 0) { cpart[r] = red[0]; dout[DO_IDX + r] = (float)idx; }
}

// ---------- decoder stage 1: ConvT1d(512,64,2,s2)+ReLU via WMMA ----------
__global__ __launch_bounds__(256) void vq_dec1(const unsigned short* __restrict__ qbf,
                                               const unsigned short* __restrict__ wA,
                                               const float* __restrict__ b1,
                                               float* __restrict__ d1) {
    const int b    = blockIdx.y;
    const int pos0 = blockIdx.x * 16;
    const int lane = threadIdx.x & 31;
    const int wave = threadIdx.x >> 5;
    const int tap  = wave >> 2;                     // kernel tap 0/1
    const int oc0  = (wave & 3) * 16;
    const int l16 = lane & 15, lhalf = lane >> 4;

    const unsigned short* arow = wA + ((size_t)tap * HID + (oc0 + l16)) * CBD;
    const unsigned short* brow = qbf + ((size_t)b * TT + pos0 + l16) * CBD + lhalf * 16;
    v8f acc = {};
    #pragma unroll
    for (int kt = 0; kt < 16; ++kt) {
        const int kb = kt * 32 + lhalf * 16;
        v8bf alo = *(const v8bf*)(arow + kb);
        v8bf ahi = *(const v8bf*)(arow + kb + 8);
        v16bf a;
        #pragma unroll
        for (int q = 0; q < 8; ++q) { a[q] = alo[q]; a[8 + q] = ahi[q]; }
        v16bf bm = *(const v16bf*)(brow + kt * 32);
        acc = __builtin_amdgcn_wmma_f32_16x16x32_bf16(false, a, false, bm,
                                                      (short)0, acc, false, false);
    }
    const int pos = pos0 + l16;
    #pragma unroll
    for (int v = 0; v < 8; ++v) {
        int oc = oc0 + lhalf * 8 + v;
        float val = acc[v] + b1[oc];
        d1[((size_t)b * HID + oc) * 4096 + 2 * pos + tap] = val > 0.f ? val : 0.f;
    }
}

// ---------- decoder stages 2+3 fused ----------
__global__ __launch_bounds__(256) void vq_dec23(const float* __restrict__ d1,
                                                const float* __restrict__ w2d,
                                                const float* __restrict__ b2d,
                                                const float* __restrict__ w3d,
                                                const float* __restrict__ b3d,
                                                float* __restrict__ dout) {
    __shared__ float w2s[HID * HID * 2];            // 32 KB
    for (int j = threadIdx.x; j < HID * HID * 2; j += 256) w2s[j] = w2d[j];
    __syncthreads();
    const int id = blockIdx.x * 256 + threadIdx.x;  // [b][o], o fastest
    const int o = id & 8191, b = id >> 13;
    const int k = o & 1, i = o >> 1;
    float col[HID];
    const float* dp = d1 + (size_t)b * HID * 4096 + i;
    #pragma unroll
    for (int ic = 0; ic < HID; ++ic) col[ic] = dp[ic * 4096];
    float acc = b3d[0];
    for (int oc = 0; oc < HID; ++oc) {
        float s = b2d[oc];
        #pragma unroll
        for (int ic = 0; ic < HID; ++ic) s += col[ic] * w2s[(ic * HID + oc) * 2 + k];
        acc += w3d[oc] * (s > 0.f ? s : 0.f);
    }
    dout[DO_OUT + id] = acc;
}

// ---------- recon loss partials ----------
__global__ __launch_bounds__(256) void vq_recon_part(const float* __restrict__ img,
                                                     const float* __restrict__ dout,
                                                     float* __restrict__ rpart) {
    __shared__ float red[256];
    float s = 0.f;
    for (int j = blockIdx.x * 256 + threadIdx.x; j < NB * LEN; j += 256 * 1024) {
        float d = img[j] - dout[DO_OUT + j];
        s += d * d;
    }
    red[threadIdx.x] = s; __syncthreads();
    for (int o = 128; o > 0; o >>= 1) { if (threadIdx.x < o) red[threadIdx.x] += red[threadIdx.x + o]; __syncthreads(); }
    if (threadIdx.x == 0) rpart[blockIdx.x] = red[0];
}

// ---------- deterministic final reduction ----------
__global__ __launch_bounds__(1024) void vq_reduce(const float* __restrict__ src, int n,
                                                  float* __restrict__ dst, float scale) {
    __shared__ float red[1024];
    float s = 0.f;
    for (int j = threadIdx.x; j < n; j += 1024) s += src[j];
    red[threadIdx.x] = s; __syncthreads();
    for (int o = 512; o > 0; o >>= 1) { if (threadIdx.x < o) red[threadIdx.x] += red[threadIdx.x + o]; __syncthreads(); }
    if (threadIdx.x == 0) dst[0] = red[0] * scale;
}

extern "C" void kernel_launch(void* const* d_in, const int* in_sizes, int n_in,
                              void* d_out, int out_size, void* d_ws, size_t ws_size,
                              hipStream_t stream) {
    (void)in_sizes; (void)n_in; (void)out_size; (void)ws_size;
    const float* img      = (const float*)d_in[0];
    const float* enc_w1   = (const float*)d_in[1];
    const float* enc_b1   = (const float*)d_in[2];
    const float* enc_w2   = (const float*)d_in[3];
    const float* enc_b2   = (const float*)d_in[4];
    const float* enc_w3   = (const float*)d_in[5];
    const float* enc_b3   = (const float*)d_in[6];
    const float* dec_w1   = (const float*)d_in[7];
    const float* dec_b1   = (const float*)d_in[8];
    const float* dec_w2   = (const float*)d_in[9];
    const float* dec_b2   = (const float*)d_in[10];
    const float* dec_w3   = (const float*)d_in[11];
    const float* dec_b3   = (const float*)d_in[12];
    const float* codebook = (const float*)d_in[13];
    float* dout = (float*)d_out;
    char*  ws   = (char*)d_ws;

    float*          flat   = (float*)(ws + OFF_FLAT);
    unsigned short* flatbf = (unsigned short*)(ws + OFF_FLATBF);
    unsigned short* qbf    = (unsigned short*)(ws + OFF_QBF);
    float*          h1     = (float*)(ws + OFF_H1);
    float*          h2     = (float*)(ws + OFF_H2);
    float*          d1     = (float*)(ws + OFF_D1);
    unsigned short* cbbf   = (unsigned short*)(ws + OFF_CBBF);
    float*          cbn    = (float*)(ws + OFF_CBN);
    unsigned short* wA     = (unsigned short*)(ws + OFF_WA);
    int*            idxb   = (int*)(ws + OFF_IDXI);
    float*          cpart  = (float*)(ws + OFF_CPART);
    float*          rpart  = (float*)(ws + OFF_RPART);

    vq_prep_cb<<<NTOK, 256, 0, stream>>>(codebook, cbbf, cbn);
    vq_prep_w <<<256, 256, 0, stream>>>(dec_w1, wA);
    vq_enc1   <<<(NB * HID * 4096) / 256, 256, 0, stream>>>(img, enc_w1, enc_b1, h1);
    vq_enc2   <<<dim3(8, HID, NB), 256, 0, stream>>>(h1, enc_w2, enc_b2, h2);
    vq_enc3   <<<dim3(NROW, 2), 256, 0, stream>>>(h2, enc_w3, enc_b3, flat, flatbf);
    vq_argmin <<<NROW / 128, 256, 0, stream>>>(flatbf, cbbf, cbn, idxb);
    vq_gather <<<NROW, 256, 0, stream>>>(idxb, codebook, flat, dout, qbf, cpart);
    vq_reduce <<<1, 1024, 0, stream>>>(cpart, NROW, dout + DO_COMMIT, 1.0f / 33554432.0f);
    vq_dec1   <<<dim3(TT / 16, NB), 256, 0, stream>>>(qbf, wA, dec_b1, d1);
    vq_dec23  <<<(NB * LEN) / 256, 256, 0, stream>>>(d1, dec_w2, dec_b2, dec_w3, dec_b3, dout);
    vq_recon_part<<<1024, 256, 0, stream>>>(img, dout, rpart);
    vq_reduce <<<1, 1024, 0, stream>>>(rpart, 1024, dout + DO_RECON, 1.0f / 262144.0f);
}
